// T5Gemma2SelfAttention_70308614635870
// MI455X (gfx1250) — matrix operand order repaired and checked by reading, not compiled, not run on hardware
//
#include <hip/hip_runtime.h>
#include <cstdint>
#include <cstddef>

#define NH   8
#define NKV  4
#define HD   256
#define HID  2048
#define BB   2
#define SS   2048

typedef __attribute__((ext_vector_type(16))) __bf16          v16bf;
typedef __attribute__((ext_vector_type(8)))  float           v8f;
typedef __attribute__((ext_vector_type(8)))  unsigned short  v8u;
typedef __attribute__((ext_vector_type(4)))  unsigned int    v4ui;
typedef __attribute__((ext_vector_type(8)))  int             v8i;
typedef __attribute__((ext_vector_type(4)))  int             v4i;

struct U16Pair { v8u lo; v8u hi; };

#if defined(__has_builtin)
#if __has_builtin(__builtin_amdgcn_tensor_load_to_lds) && __has_builtin(__builtin_amdgcn_s_wait_tensorcnt)
#define HAVE_TDM 1
#endif
#endif

static __device__ __forceinline__ unsigned short f2bf(float f) {
  unsigned u = __builtin_bit_cast(unsigned, f);
  u += 0x7FFFu + ((u >> 16) & 1u);               // round-to-nearest-even
  return (unsigned short)(u >> 16);
}

// Load a 16-element bf16 fragment from two 16-byte LDS chunks.
static __device__ __forceinline__ v16bf frag_ld(const unsigned short* p0,
                                                const unsigned short* p1) {
  U16Pair t;
  t.lo = *(const v8u*)p0;
  t.hi = *(const v8u*)p1;
  return __builtin_bit_cast(v16bf, t);
}

#ifdef HAVE_TDM
// TDM: load a 2D bf16 tile (tile_rows x 256 elems) from global into LDS with
// an 8-ushort (4-DWORD) pad after each 256-elem (128-DWORD) row -> stride 264.
// D# bitfields per CDNA5 ISA 8.3/8.4. rows_avail clamps OOB rows to zero-fill.
static __device__ __forceinline__ void tdm_load_row_tile(
    unsigned lds_byte_addr, const unsigned short* gptr,
    unsigned rows_avail, unsigned tile_rows) {
  unsigned long long ga = (unsigned long long)(uintptr_t)gptr;
  v4ui g0;
  g0.x = 1u;                                                 // count=1, user desc
  g0.y = lds_byte_addr;                                      // lds_addr
  g0.z = (unsigned)(ga & 0xFFFFFFFFu);                       // global_addr[31:0]
  g0.w = (unsigned)((ga >> 32) & 0x01FFFFFFu) | (2u << 30);  // addr[56:32] | type=2
  v8i g1;
  g1[0] = (int)((1u << 16) |                                 // data_size = 2B
                (1u << 20) |                                 // pad_enable
                (6u << 22) |                                 // pad_interval: 128 DW
                (3u << 25));                                 // pad_amount: 4 DW
  g1[1] = (int)((unsigned)HD << 16);                         // tensor_dim0 = 256
  g1[2] = (int)((rows_avail & 0xFFFFu) << 16);               // tensor_dim1 lo
  g1[3] = (int)(((rows_avail >> 16) & 0xFFFFu) |
                ((unsigned)HD << 16));                       // dim1 hi | tile_dim0
  g1[4] = (int)(tile_rows & 0xFFFFu);                        // tile_dim1, tile_dim2=0
  g1[5] = (int)HD;                                           // tensor_dim0_stride
  g1[6] = 0;
  g1[7] = 0;
  v4i z4 = {0, 0, 0, 0};
#if defined(__clang_major__) && (__clang_major__ >= 23)
  v8i z8 = {0, 0, 0, 0, 0, 0, 0, 0};
  __builtin_amdgcn_tensor_load_to_lds(g0, g1, z4, z4, z8, 0);
#else
  __builtin_amdgcn_tensor_load_to_lds(g0, g1, z4, z4, 0);
#endif
}
#endif

// ---------------------------------------------------------------------------
// Tiled WMMA GEMM: C(f32) = A(f32 or bf16) * B(f32), 128x64 tile, BK=32.
// 256 threads = 8 waves (4x2), each wave a 32x32 sub-tile (2x2 WMMA frags).
// ---------------------------------------------------------------------------
template<bool ABF16>
__global__ __launch_bounds__(256) void gemm_bf16_wmma(
    const void* __restrict__ Av, const float* __restrict__ Bm,
    float* __restrict__ Cm, int M, int N, int K) {
  __shared__ unsigned short As[128 * 40];  // [m][k], stride 40 (pad 8)
  __shared__ unsigned short Bt[64 * 40];   // [n][k], stride 40 (pad 8)

  const int tid  = threadIdx.x;
  const int lane = tid & 31;
  const int w    = tid >> 5;
  const int half = lane >> 4;
  const int ln   = lane & 15;
  const int wm   = (w >> 1) * 32;          // 0,32,64,96
  const int wn   = (w & 1) * 32;           // 0,32
  const int m0   = blockIdx.y * 128;
  const int n0   = blockIdx.x * 64;
  const float*          Af = (const float*)Av;
  const unsigned short* Ab = (const unsigned short*)Av;

  v8f acc[2][2];
  #pragma unroll
  for (int i = 0; i < 2; i++)
    #pragma unroll
    for (int j = 0; j < 2; j++) acc[i][j] = {};

  for (int k0 = 0; k0 < K; k0 += 32) {
    __syncthreads();
    if (k0 + 32 < K) {                     // global_prefetch_b8 of next K-tile
      int pr = tid >> 1, pc = (tid & 1) * 16;
      if (ABF16) __builtin_prefetch(&Ab[(size_t)(m0 + pr) * K + k0 + 32 + pc], 0, 1);
      else       __builtin_prefetch(&Af[(size_t)(m0 + pr) * K + k0 + 32 + pc], 0, 1);
      int br = tid >> 3, bc = (tid & 7) * 8;
      __builtin_prefetch(&Bm[(size_t)(k0 + 32 + br) * N + n0 + bc], 0, 1);
    }
    // Stage A (128x32) as bf16
    #pragma unroll
    for (int i = 0; i < 16; i++) {
      int idx = i * 256 + tid;
      int r = idx >> 5, c = idx & 31;
      unsigned short av;
      if (ABF16) av = Ab[(size_t)(m0 + r) * K + k0 + c];
      else       av = f2bf(Af[(size_t)(m0 + r) * K + k0 + c]);
      As[r * 40 + c] = av;
    }
    // Stage B (32x64) transposed as bf16
    #pragma unroll
    for (int i = 0; i < 8; i++) {
      int idx = i * 256 + tid;
      int rk = idx >> 6, cn = idx & 63;
      Bt[cn * 40 + rk] = f2bf(Bm[(size_t)(k0 + rk) * N + n0 + cn]);
    }
    __syncthreads();

    const int b0  = half * 8;
    const int kbh = half * 16;
    v16bf a0 = frag_ld(&As[(wm      + ln) * 40 + kbh], &As[(wm      + ln) * 40 + kbh + 8]);
    v16bf a1 = frag_ld(&As[(wm + 16 + ln) * 40 + kbh], &As[(wm + 16 + ln) * 40 + kbh + 8]);
    v16bf bf0 = frag_ld(&Bt[(wn      + ln) * 40 + kbh], &Bt[(wn      + ln) * 40 + kbh + 8]);
    v16bf bf1 = frag_ld(&Bt[(wn + 16 + ln) * 40 + kbh], &Bt[(wn + 16 + ln) * 40 + kbh + 8]);
    acc[0][0] = __builtin_amdgcn_wmma_f32_16x16x32_bf16(false, a0, false, bf0, (short)0, acc[0][0], false, false);
    acc[0][1] = __builtin_amdgcn_wmma_f32_16x16x32_bf16(false, a0, false, bf1, (short)0, acc[0][1], false, false);
    acc[1][0] = __builtin_amdgcn_wmma_f32_16x16x32_bf16(false, a1, false, bf0, (short)0, acc[1][0], false, false);
    acc[1][1] = __builtin_amdgcn_wmma_f32_16x16x32_bf16(false, a1, false, bf1, (short)0, acc[1][1], false, false);
  }

  #pragma unroll
  for (int fi = 0; fi < 2; fi++)
    #pragma unroll
    for (int fj = 0; fj < 2; fj++)
      #pragma unroll
      for (int j = 0; j < 8; j++) {
        int r = m0 + wm + fi * 16 + j + half * 8;
        int c = n0 + wn + fj * 16 + ln;
        Cm[(size_t)r * N + c] = acc[fi][fj][j];
      }
}

// ---------------------------------------------------------------------------
// Gemma RMSNorm + RoPE (q,k) / bf16 convert (v). One block per (b,s,head).
// ---------------------------------------------------------------------------
__global__ __launch_bounds__(256) void rmsnorm_rope_kernel(
    const float* __restrict__ qraw, const float* __restrict__ kraw,
    const float* __restrict__ vraw, const float* __restrict__ cosT,
    const float* __restrict__ sinT, const float* __restrict__ qw,
    const float* __restrict__ kw, unsigned short* __restrict__ qbf,
    unsigned short* __restrict__ kbf, unsigned short* __restrict__ vbf) {
  __shared__ float red[HD];
  __shared__ float arr[HD];
  const int row = blockIdx.x;            // b*S + s
  const int b   = row >> 11;
  const int s   = row & (SS - 1);
  const int hy  = blockIdx.y;
  const int t   = threadIdx.x;

  if (hy >= NH + NKV) {                  // V heads: convert only
    int h = hy - (NH + NKV);
    float x = vraw[(size_t)row * (NKV * HD) + h * HD + t];
    vbf[((size_t)(b * NKV + h) * SS + s) * HD + t] = f2bf(x);
    return;
  }

  float x; const float* wv;
  if (hy < NH) { x = qraw[(size_t)row * (NH * HD) + hy * HD + t]; wv = qw; }
  else { int h = hy - NH; x = kraw[(size_t)row * (NKV * HD) + h * HD + t]; wv = kw; }

  red[t] = x * x;
  __syncthreads();
  for (int off = 128; off > 0; off >>= 1) {
    if (t < off) red[t] += red[t + off];
    __syncthreads();
  }
  float inv = rsqrtf(red[0] * (1.0f / HD) + 1e-6f);
  float xn  = x * inv * (1.0f + wv[t]);
  arr[t] = xn;
  __syncthreads();
  float rot = (t < HD / 2) ? -arr[t + HD / 2] : arr[t - HD / 2];
  float cs = cosT[(size_t)row * HD + t];
  float sn = sinT[(size_t)row * HD + t];
  float o = xn * cs + rot * sn;
  if (hy < NH) qbf[((size_t)(b * NH + hy) * SS + s) * HD + t] = f2bf(o);
  else { int h = hy - NH; kbf[((size_t)(b * NKV + h) * SS + s) * HD + t] = f2bf(o); }
}

// ---------------------------------------------------------------------------
// Sliding-window flash attention, WMMA bf16. 64 q rows per block, 8 waves:
// wave = (qsub 0..3) x (role 0..1). Each role-wave computes one 16-col score
// group over full d, and accumulates one 128-wide d-half for PV.
// Row stats are exchanged between the role pair via LDS.
// ---------------------------------------------------------------------------
__global__ __launch_bounds__(256) void attn_kernel(
    const unsigned short* __restrict__ qbf, const unsigned short* __restrict__ kbf,
    const unsigned short* __restrict__ vbf, unsigned short* __restrict__ obf) {
  extern __shared__ unsigned short sm[];
  unsigned short* Qs = sm;                 // 64 x 264
  unsigned short* Ks = Qs + 64 * 264;      // 32 x 264
  unsigned short* Vt = Ks + 32 * 264;      // 256 x 40 (transposed V)
  unsigned short* Ps = Vt + 256 * 40;      // 4 qsub x 16 x 40 (P bounce)
  float* Rm = (float*)(Ps + 4 * 16 * 40);  // 4 qsub x 2 grp x 16 rows (row max)
  float* Sm = Rm + 128;                    // 4 qsub x 2 grp x 16 rows (row sum)

  const int tid  = threadIdx.x;
  const int lane = tid & 31;
  const int w    = tid >> 5;
  const int qsub = w & 3;
  const int role = w >> 2;                 // 0/1: score group & d-half
  const int half = lane >> 4;
  const int ln   = lane & 15;
  const int q0   = blockIdx.x * 64;
  const int bh   = blockIdx.y;
  const int b    = bh >> 3;
  const int h    = bh & 7;
  const int kvh  = h >> 1;                 // GQA repeat=2
  const size_t qbase  = (size_t)(b * NH  + h)   * SS * HD;
  const size_t kvbase = (size_t)(b * NKV + kvh) * SS * HD;
  const int qrow0 = q0 + qsub * 16;
  const int b0  = half * 8;
  const int kbh = half * 16;
  const v8u ZV8 = {0, 0, 0, 0, 0, 0, 0, 0};

  // --- Stage Q tile (64 x 256 -> stride 264) once ---
#ifdef HAVE_TDM
  if (tid < 32) {
    tdm_load_row_tile((unsigned)(uintptr_t)Qs, qbf + qbase + (size_t)q0 * HD,
                      (unsigned)(SS - q0), 64u);
    __builtin_amdgcn_s_wait_tensorcnt(0);
  }
  __asm__ volatile("" ::: "memory");
#else
  #pragma unroll
  for (int i = 0; i < 8; i++) {
    int ch = i * 256 + tid;
    int r = ch >> 5, off = (ch & 31) * 8;
    *(v8u*)&Qs[r * 264 + off] = *(const v8u*)&qbf[qbase + (size_t)(q0 + r) * HD + off];
  }
#endif
  __syncthreads();

  v8f acc[8];
  #pragma unroll
  for (int i = 0; i < 8; i++) acc[i] = {};
  float mrow[8], lrow[8];
  #pragma unroll
  for (int j = 0; j < 8; j++) { mrow[j] = -3.0e38f; lrow[j] = 0.0f; }

  int klo = q0 - 511; if (klo < 0) klo = 0;
  int kb0 = klo & ~31;
  int khi = q0 + 63 + 512; if (khi > SS - 1) khi = SS - 1;

  for (int kb = kb0; kb <= khi; kb += 32) {
    __syncthreads();
    // --- Stage K (32 x 256 -> stride 264) ---
#ifdef HAVE_TDM
    if (tid < 32) {
      tdm_load_row_tile((unsigned)(uintptr_t)Ks, kbf + kvbase + (size_t)kb * HD,
                        (unsigned)(SS - kb), 32u);
      __builtin_amdgcn_s_wait_tensorcnt(0);
    }
    __asm__ volatile("" ::: "memory");
#else
    #pragma unroll
    for (int i = 0; i < 4; i++) {
      int ch = i * 256 + tid;
      int r = ch >> 5, off = (ch & 31) * 8;
      if (kb + r < SS)
        *(v8u*)&Ks[r * 264 + off] = *(const v8u*)&kbf[kvbase + (size_t)(kb + r) * HD + off];
      else
        *(v8u*)&Ks[r * 264 + off] = ZV8;
    }
#endif
    // --- Stage V transposed: Vt[d][kk] ---
    #pragma unroll
    for (int i = 0; i < 32; i++) {
      int e = i * 256 + tid;
      int kk = e >> 8, d = e & 255;
      unsigned short val = 0;
      if (kb + kk < SS) val = vbf[kvbase + (size_t)(kb + kk) * HD + d];
      Vt[d * 40 + kk] = val;
    }
    __syncthreads();

    // --- Phase A: scores for this wave's 16-col group (full d contraction) ---
    v8f sc = {};
    #pragma unroll
    for (int dc = 0; dc < 8; dc++) {
      v16bf a = frag_ld(&Qs[(qsub * 16 + ln) * 264 + dc * 32 + kbh],
                        &Qs[(qsub * 16 + ln) * 264 + dc * 32 + kbh + 8]);
      v16bf kf = frag_ld(&Ks[(role * 16 + ln) * 264 + dc * 32 + kbh],
                         &Ks[(role * 16 + ln) * 264 + dc * 32 + kbh + 8]);
      sc = __builtin_amdgcn_wmma_f32_16x16x32_bf16(false, a, false, kf, (short)0, sc, false, false);
    }
    float rmax[8];
    #pragma unroll
    for (int j = 0; j < 8; j++) {
      float sv = sc[j] * 0.0625f;                    // 256^-0.5
      sv = tanhf(sv * (1.0f / 50.0f)) * 50.0f;       // softcap
      int qi = qrow0 + j + half * 8;
      int ki = kb + role * 16 + ln;
      int dist = qi - ki;
      bool valid = (dist >= -512) && (dist <= 511) && (ki < SS);
      sv = valid ? sv : -3.0e38f;
      sc[j] = sv;
      rmax[j] = sv;
    }
    #pragma unroll
    for (int j = 0; j < 8; j++)
      #pragma unroll
      for (int o = 1; o < 16; o <<= 1)
        rmax[j] = fmaxf(rmax[j], __shfl_xor(rmax[j], o, 32));
    if (ln == 0) {
      #pragma unroll
      for (int j = 0; j < 8; j++) Rm[qsub * 32 + role * 16 + half * 8 + j] = rmax[j];
    }
    __syncthreads();

    // --- Phase B: combine row max across groups, P = exp(s - m) ---
    float mnew[8], scalef[8], psum[8];
    #pragma unroll
    for (int j = 0; j < 8; j++) {
      int row = half * 8 + j;
      float rt = fmaxf(Rm[qsub * 32 + row], Rm[qsub * 32 + 16 + row]);
      mnew[j] = fmaxf(mrow[j], rt);
      scalef[j] = __expf(mrow[j] - mnew[j]);
      mrow[j] = mnew[j];
      psum[j] = 0.0f;
    }
    unsigned short* Pw = Ps + qsub * 16 * 40;
    #pragma unroll
    for (int j = 0; j < 8; j++) {
      int qi = qrow0 + j + half * 8;
      int ki = kb + role * 16 + ln;
      int dist = qi - ki;
      bool valid = (dist >= -512) && (dist <= 511) && (ki < SS);
      float p = valid ? __expf(sc[j] - mnew[j]) : 0.0f;
      psum[j] = p;
      Pw[(j + half * 8) * 40 + role * 16 + ln] = f2bf(p);
    }
    #pragma unroll
    for (int j = 0; j < 8; j++)
      #pragma unroll
      for (int o = 1; o < 16; o <<= 1)
        psum[j] += __shfl_xor(psum[j], o, 32);
    if (ln == 0) {
      #pragma unroll
      for (int j = 0; j < 8; j++) Sm[qsub * 32 + role * 16 + half * 8 + j] = psum[j];
    }
    __syncthreads();

    // --- Phase C: rescale, PV over this wave's d-half ---
    #pragma unroll
    for (int j = 0; j < 8; j++) {
      int row = half * 8 + j;
      float pt = Sm[qsub * 32 + row] + Sm[qsub * 32 + 16 + row];
      lrow[j] = lrow[j] * scalef[j] + pt;
    }
    #pragma unroll
    for (int i = 0; i < 8; i++)
      #pragma unroll
      for (int j = 0; j < 8; j++) acc[i][j] *= scalef[j];
    v16bf pa = frag_ld(&Pw[ln * 40 + kbh], &Pw[ln * 40 + kbh + 8]);
    #pragma unroll
    for (int i = 0; i < 8; i++) {
      int dcg = role * 8 + i;
      v16bf vf = frag_ld(&Vt[(dcg * 16 + ln) * 40 + kbh],
                         &Vt[(dcg * 16 + ln) * 40 + kbh + 8]);
      acc[i] = __builtin_amdgcn_wmma_f32_16x16x32_bf16(false, pa, false, vf, (short)0, acc[i], false, false);
    }
  }

  // Epilogue: normalize, write this wave's d-half as bf16 for the Wo GEMM
  #pragma unroll
  for (int i = 0; i < 8; i++)
    #pragma unroll
    for (int j = 0; j < 8; j++) {
      int qi = qrow0 + j + half * 8;
      float o = acc[i][j] / lrow[j];
      obf[(size_t)(b * SS + qi) * (NH * HD) + h * HD + role * 128 + i * 16 + ln] = f2bf(o);
    }
}

// ---------------------------------------------------------------------------
extern "C" void kernel_launch(void* const* d_in, const int* in_sizes, int n_in,
                              void* d_out, int out_size, void* d_ws, size_t ws_size,
                              hipStream_t stream) {
  (void)in_sizes; (void)n_in; (void)out_size; (void)ws_size;
  const float* X    = (const float*)d_in[0];
  const float* Wq   = (const float*)d_in[1];
  const float* Wk   = (const float*)d_in[2];
  const float* Wv   = (const float*)d_in[3];
  const float* Wo   = (const float*)d_in[4];
  const float* qw   = (const float*)d_in[5];
  const float* kw   = (const float*)d_in[6];
  const float* cosT = (const float*)d_in[7];
  const float* sinT = (const float*)d_in[8];

  char* ws = (char*)d_ws;
  size_t off = 0;
  float* qraw = (float*)(ws + off);          off += (size_t)BB * SS * NH  * HD * 4;
  float* kraw = (float*)(ws + off);          off += (size_t)BB * SS * NKV * HD * 4;
  float* vraw = (float*)(ws + off);          off += (size_t)BB * SS * NKV * HD * 4;
  unsigned short* qbf = (unsigned short*)(ws + off); off += (size_t)BB * SS * NH  * HD * 2;
  unsigned short* kbf = (unsigned short*)(ws + off); off += (size_t)BB * SS * NKV * HD * 2;
  unsigned short* vbf = (unsigned short*)(ws + off); off += (size_t)BB * SS * NKV * HD * 2;
  unsigned short* obf = (unsigned short*)(ws + off); off += (size_t)BB * SS * NH  * HD * 2;

  const int M = BB * SS;
  gemm_bf16_wmma<false><<<dim3((NH  * HD) / 64, M / 128), 256, 0, stream>>>(X, Wq, qraw, M, NH  * HD, HID);
  gemm_bf16_wmma<false><<<dim3((NKV * HD) / 64, M / 128), 256, 0, stream>>>(X, Wk, kraw, M, NKV * HD, HID);
  gemm_bf16_wmma<false><<<dim3((NKV * HD) / 64, M / 128), 256, 0, stream>>>(X, Wv, vraw, M, NKV * HD, HID);

  rmsnorm_rope_kernel<<<dim3(M, NH + 2 * NKV), 256, 0, stream>>>(
      qraw, kraw, vraw, cosT, sinT, qw, kw, qbf, kbf, vbf);

  size_t shm = (size_t)(64 * 264 + 32 * 264 + 256 * 40 + 4 * 16 * 40) * 2 + 2 * 128 * 4;
  attn_kernel<<<dim3(SS / 64, BB * NH), 256, shm, stream>>>(qbf, kbf, vbf, obf);

  gemm_bf16_wmma<true><<<dim3(HID / 64, M / 128), 256, 0, stream>>>(obf, Wo, (float*)d_out, M, HID, HID);
}
